// SelectToken_17471926960480
// MI455X (gfx1250) — matrix operand order repaired and checked by reading, not compiled, not run on hardware
//
#include <hip/hip_runtime.h>
#include <hip/hip_bf16.h>

// ---- problem constants (match reference) ----
#define B_SZ  32
#define NT    256
#define NS    4096
#define C_DIM 256
#define NWIN  64
#define WSA   64
#define TOPK_ 32

// LDS row stride in bf16 elements: 256 + 8 pad -> 528B row stride, keeps the
// 16 lanes of an A-fragment load on distinct bank groups (528/4 = 132 == 4 mod 64,
// each lane reads 16B = 4 banks -> conflict-free ds_load_b128).
#define LROW 264

typedef __attribute__((ext_vector_type(16))) __bf16 bf16x16;
typedef __attribute__((ext_vector_type(8)))  __bf16 bf16x8;
typedef __attribute__((ext_vector_type(8)))  float  f32x8;

// ---------------------------------------------------------------------------
// 1) z_max over Nt
// ---------------------------------------------------------------------------
__global__ __launch_bounds__(256)
void stk_zmax_kernel(const float* __restrict__ z, float* __restrict__ zmax) {
    const int b = blockIdx.x;
    const int c = threadIdx.x;
    float m = -3.402823466e38f;
    for (int n = 0; n < NT; ++n)
        m = fmaxf(m, z[((size_t)b * NT + n) * C_DIM + c]);
    zmax[b * C_DIM + c] = m;
}

// ---------------------------------------------------------------------------
// 2) window scores: mean over window of dot(z_max, x_token)/C
//    = dot(z_max, sum_tokens x)/ (C*64). One block per (b, window).
// ---------------------------------------------------------------------------
__global__ __launch_bounds__(256)
void stk_wscore_kernel(const float* __restrict__ x, const float* __restrict__ zmax,
                       float* __restrict__ wscore) {
    __shared__ float red[256];
    const int b  = blockIdx.x >> 6;
    const int w  = blockIdx.x & 63;
    const int wr = w >> 3, wc = w & 7;
    const int c  = threadIdx.x;
    float acc = 0.0f;
    for (int r = 0; r < WSA; ++r) {
        const int i = r >> 3, j = r & 7;
        const int tok = (wr * 8 + i) * 64 + (wc * 8 + j);
        acc += x[((size_t)b * NS + tok) * C_DIM + c];
    }
    acc *= zmax[b * C_DIM + c];
    red[c] = acc;
    __syncthreads();
    for (int s = 128; s > 0; s >>= 1) {
        if (c < s) red[c] += red[c + s];
        __syncthreads();
    }
    if (c == 0) wscore[b * NWIN + w] = red[0] * (1.0f / (C_DIM * WSA));
}

// ---------------------------------------------------------------------------
// 3) top-32 of 64 by rank counting (stable: descending value, lower idx first,
//    matching jax.lax.top_k ordering).
// ---------------------------------------------------------------------------
__global__ __launch_bounds__(64)
void stk_topk_kernel(const float* __restrict__ wscore, int* __restrict__ idxbuf) {
    __shared__ float s[NWIN];
    const int b = blockIdx.x;
    const int t = threadIdx.x;
    s[t] = wscore[b * NWIN + t];
    __syncthreads();
    const float mine = s[t];
    int rank = 0;
    for (int u = 0; u < NWIN; ++u) {
        const float v = s[u];
        rank += (v > mine) || (v == mine && u < t);
    }
    if (rank < TOPK_) idxbuf[b * TOPK_ + rank] = t;
}

// ---------------------------------------------------------------------------
// 4) convert Wd, Wu (f32 row-major [out_ch][in_ch]) to bf16 in workspace
// ---------------------------------------------------------------------------
__global__ __launch_bounds__(256)
void stk_wconv_kernel(const float* __restrict__ Wd, const float* __restrict__ Wu,
                      __bf16* __restrict__ WdB, __bf16* __restrict__ WuB) {
    const int id = blockIdx.x * 256 + threadIdx.x;
    if (id < C_DIM * C_DIM) WdB[id] = (__bf16)Wd[id];
    else                    WuB[id - C_DIM * C_DIM] = (__bf16)Wu[id - C_DIM * C_DIM];
}

// ---------------------------------------------------------------------------
// 5) fused: gather window -> GEMM1 (Wd) -> shift -> GEMM2 (Wu) -> +xe -> out
//    One block per (b, k). 256 threads = 8 waves; each wave owns a
//    16-row x 128-col output tile (8 WMMA accumulators of 16x16).
// ---------------------------------------------------------------------------
__global__ __launch_bounds__(256)
void stk_fused_mlp_kernel(const float* __restrict__ x,
                          const int* __restrict__ idxbuf,
                          const __bf16* __restrict__ WdB,
                          const __bf16* __restrict__ WuB,
                          const float* __restrict__ bd,
                          const float* __restrict__ bu,
                          float* __restrict__ out)
{
    extern __shared__ char smem[];
    __bf16* XE   = (__bf16*)smem;        // [64][LROW] gathered window, bf16
    __bf16* Tbuf = XE + 64 * LROW;       // [64][LROW] t = xe@Wd^T + bd, bf16

    const int b    = blockIdx.x >> 5;
    const int ksel = blockIdx.x & 31;
    const int w    = idxbuf[b * TOPK_ + ksel];
    const int wr = w >> 3, wc = w & 7;

    const int tid  = threadIdx.x;
    const int lane = tid & 31;
    const int wave = tid >> 5;
    const int l16  = lane & 15;
    const int hi   = lane >> 4;           // lane half (K-half selector)
    const int tm   = wave >> 1;           // row tile 0..3 (16 rows each)
    const int colbase = (wave & 1) * 128; // column half

    // ---- stage gathered window into LDS (f32 -> bf16) ----
    {
        const int rsub = tid >> 6;        // 0..3
        const int c0   = (tid & 63) * 4;
        #pragma unroll
        for (int rg = 0; rg < 16; ++rg) {
            const int r = rg * 4 + rsub;
            const int i = r >> 3, j = r & 7;
            const int tok = (wr * 8 + i) * 64 + (wc * 8 + j);
            const float4 v = *(const float4*)&x[((size_t)b * NS + tok) * C_DIM + c0];
            XE[r * LROW + c0 + 0] = (__bf16)v.x;
            XE[r * LROW + c0 + 1] = (__bf16)v.y;
            XE[r * LROW + c0 + 2] = (__bf16)v.z;
            XE[r * LROW + c0 + 3] = (__bf16)v.w;
        }
    }
    __syncthreads();

    const int mrow = tm * 16 + l16;       // this lane's A-fragment row

    // ---- GEMM1: t = xe @ Wd^T ----
    f32x8 acc[8];
    #pragma unroll
    for (int nt = 0; nt < 8; ++nt)
        #pragma unroll
        for (int e = 0; e < 8; ++e) acc[nt][e] = 0.0f;

    #pragma unroll
    for (int ks = 0; ks < 8; ++ks) {
        // A fragment (16x32 bf16): elems 0..7 -> K = ks*32+hi*8+(0..7),
        //                          elems 8..15 -> +16 (per ISA A layout)
        const int ka = ks * 32 + hi * 8;
        bf16x8 a0 = *(const bf16x8*)&XE[mrow * LROW + ka];
        bf16x8 a1 = *(const bf16x8*)&XE[mrow * LROW + ka + 16];
        bf16x16 A;
        #pragma unroll
        for (int e = 0; e < 8; ++e) { A[e] = a0[e]; A[e + 8] = a1[e]; }

        #pragma unroll
        for (int nt = 0; nt < 8; ++nt) {
            // B fragment (32x16): lane holds B column n = weight row n,
            // K = ks*32 + hi*8 + (0..7), then +16 (same order as A)
            const int n = colbase + nt * 16 + l16;
            const __bf16* bp = WdB + n * C_DIM + ks * 32 + hi * 8;
            __builtin_prefetch((const void*)(bp + C_DIM), 0, 1);
            bf16x8 b0 = *(const bf16x8*)bp;
            bf16x8 b1 = *(const bf16x8*)(bp + 16);
            bf16x16 Bf;
            #pragma unroll
            for (int e = 0; e < 8; ++e) { Bf[e] = b0[e]; Bf[e + 8] = b1[e]; }
            acc[nt] = __builtin_amdgcn_wmma_f32_16x16x32_bf16(
                false, A, false, Bf, (short)0, acc[nt], false, false);
        }
    }

    // ---- t + bd -> LDS bf16 (C/D layout: M = tm*16 + 8*hi + r, N = l16) ----
    #pragma unroll
    for (int nt = 0; nt < 8; ++nt) {
        const int n = colbase + nt * 16 + l16;
        const float bias = bd[n];
        #pragma unroll
        for (int r = 0; r < 8; ++r) {
            const int m = tm * 16 + hi * 8 + r;
            Tbuf[m * LROW + n] = (__bf16)(acc[nt][r] + bias);
        }
    }
    __syncthreads();

    // ---- GEMM2: up = shift(t) @ Wu^T ; shift applied at A-read time ----
    #pragma unroll
    for (int nt = 0; nt < 8; ++nt)
        #pragma unroll
        for (int e = 0; e < 8; ++e) acc[nt][e] = 0.0f;

    const int ii = mrow >> 3, jj = mrow & 7;   // spatial (height, width)
    #pragma unroll
    for (int ks = 0; ks < 8; ++ks) {
        bf16x16 A;
        #pragma unroll
        for (int half = 0; half < 2; ++half) {
            const int kc = ks * 32 + hi * 8 + half * 16;  // 8-wide chunk, inside one group
            const int g  = kc >> 6;                        // channel group 0..3
            const int di = (g == 2) - (g == 3);            // g2: +1 height, g3: -1
            const int dj = (g == 0) - (g == 1);            // g0: +1 width,  g1: -1
            const int ni = ii + di, nj = jj + dj;
            bf16x8 av;
            if ((unsigned)ni < 8u && (unsigned)nj < 8u) {
                av = *(const bf16x8*)&Tbuf[(ni * 8 + nj) * LROW + kc];
            } else {
                #pragma unroll
                for (int e = 0; e < 8; ++e) av[e] = (__bf16)0.0f;  // zero pad
            }
            #pragma unroll
            for (int e = 0; e < 8; ++e) A[half * 8 + e] = av[e];
        }
        #pragma unroll
        for (int nt = 0; nt < 8; ++nt) {
            const int n = colbase + nt * 16 + l16;
            const __bf16* bp = WuB + n * C_DIM + ks * 32 + hi * 8;
            bf16x8 b0 = *(const bf16x8*)bp;
            bf16x8 b1 = *(const bf16x8*)(bp + 16);
            bf16x16 Bf;
            #pragma unroll
            for (int e = 0; e < 8; ++e) { Bf[e] = b0[e]; Bf[e + 8] = b1[e]; }
            acc[nt] = __builtin_amdgcn_wmma_f32_16x16x32_bf16(
                false, A, false, Bf, (short)0, acc[nt], false, false);
        }
    }

    // ---- epilogue: out = xe(f32, re-read from L2) + up + bu ----
    #pragma unroll
    for (int nt = 0; nt < 8; ++nt) {
        const int n = colbase + nt * 16 + l16;
        const float bias = bu[n];
        #pragma unroll
        for (int r = 0; r < 8; ++r) {
            const int m = tm * 16 + hi * 8 + r;
            const int i = m >> 3, j = m & 7;
            const int tok = (wr * 8 + i) * 64 + (wc * 8 + j);
            const float xev = x[((size_t)b * NS + tok) * C_DIM + n];
            out[((size_t)b * 2048 + ksel * WSA + m) * C_DIM + n] =
                acc[nt][r] + bias + xev;
        }
    }
}

// ---------------------------------------------------------------------------
extern "C" void kernel_launch(void* const* d_in, const int* in_sizes, int n_in,
                              void* d_out, int out_size, void* d_ws, size_t ws_size,
                              hipStream_t stream) {
    (void)in_sizes; (void)n_in; (void)out_size; (void)ws_size;
    const float* z  = (const float*)d_in[0];
    const float* x  = (const float*)d_in[1];
    const float* Wd = (const float*)d_in[2];
    const float* bd = (const float*)d_in[3];
    const float* Wu = (const float*)d_in[4];
    const float* bu = (const float*)d_in[5];
    float* out = (float*)d_out;

    // workspace layout (307,200 bytes total)
    char* ws = (char*)d_ws;
    float*  zmax   = (float*)ws;                       // 32*256 f32  = 32768 B
    float*  wscore = (float*)(ws + 32768);             // 32*64  f32  =  8192 B
    int*    idxbuf = (int*)  (ws + 40960);             // 32*32  i32  =  4096 B
    __bf16* WdB    = (__bf16*)(ws + 45056);            // 65536 bf16  = 131072 B
    __bf16* WuB    = WdB + C_DIM * C_DIM;              // 65536 bf16

    stk_zmax_kernel  <<<B_SZ,        256, 0, stream>>>(z, zmax);
    stk_wscore_kernel<<<B_SZ * NWIN, 256, 0, stream>>>(x, zmax, wscore);
    stk_topk_kernel  <<<B_SZ,        64,  0, stream>>>(wscore, idxbuf);
    stk_wconv_kernel <<<512,         256, 0, stream>>>(Wd, Wu, WdB, WuB);

    const size_t lds_bytes = (size_t)2 * 64 * LROW * sizeof(__bf16); // 67,584 B
    stk_fused_mlp_kernel<<<B_SZ * TOPK_, 256, lds_bytes, stream>>>(
        x, idxbuf, WdB, WuB, bd, bu, out);
}
